// QueryAwareGNN_30709016166583
// MI455X (gfx1250) — matrix operand order repaired and checked by reading, not compiled, not run on hardware
//
#include <hip/hip_runtime.h>
#include <hip/hip_bf16.h>

typedef __attribute__((ext_vector_type(16))) __bf16       v16bf;
typedef __attribute__((ext_vector_type(2)))  __bf16       bf16x2;
typedef __attribute__((ext_vector_type(8)))  float        v8f;
typedef __attribute__((ext_vector_type(2)))  float        f32x2;
typedef __attribute__((ext_vector_type(4)))  unsigned int u32x4;
typedef __attribute__((ext_vector_type(2)))  unsigned int u32x2;

#define DH 128
#define BM 128
#define BK 32
#define LDP 20   // LDS row stride in dwords: 16 pairs + pad (16B-aligned, conflict-free)

// pack two fp32 into one dword of two bf16 (lo -> [15:0]); lowers to v_cvt_pk_bf16_f32
static __device__ __forceinline__ unsigned int pk(float lo, float hi) {
  f32x2 f = {lo, hi};
  union { bf16x2 b; unsigned int u; } u;
  u.b = __builtin_convertvector(f, bf16x2);
  return u.u;
}

// ---------------- norm precompute ----------------

__global__ __launch_bounds__(256) void k_deg_init(float* deg, int n) {
  int i = blockIdx.x * blockDim.x + threadIdx.x;
  if (i < n) deg[i] = 1.0f;  // self-loop weight
}

__global__ __launch_bounds__(256) void k_deg_accum(const int* __restrict__ ei,
                                                   const float* __restrict__ w,
                                                   float* deg, int E) {
  int e = blockIdx.x * blockDim.x + threadIdx.x;
  if (e < E) atomicAdd(&deg[ei[E + e]], w[e]);
}

__global__ __launch_bounds__(256) void k_dinv(float* deg, float* selfn, int n) {
  int i = blockIdx.x * blockDim.x + threadIdx.x;
  if (i < n) {
    float d = deg[i];
    float r = d > 0.0f ? rsqrtf(d) : 0.0f;
    deg[i] = r;          // deg buffer becomes dinv
    selfn[i] = r * r;    // self-loop norm
  }
}

__global__ __launch_bounds__(256) void k_norm(const int* __restrict__ ei,
                                              const float* __restrict__ w,
                                              const float* __restrict__ dinv,
                                              float* __restrict__ normv, int E) {
  int e = blockIdx.x * blockDim.x + threadIdx.x;
  if (e < E) normv[e] = dinv[ei[e]] * w[e] * dinv[ei[E + e]];
}

// ---------------- WMMA bf16 GEMM: out[M,128] = A[M,K] @ W[K,128] (+bias) ----------------
// 256 threads = 8 waves. LDS tiles stored PRE-PACKED in the WMMA pair layout so
// fragment builds are 2x ds_load_b128 each. Register double-buffer on global fetch.

__global__ __launch_bounds__(256, 2)
void k_gemm_wmma_bf16(const float* __restrict__ A, const float* __restrict__ W,
                      const float* __restrict__ bias, float* __restrict__ out,
                      int M, int K) {
  __shared__ unsigned int sAp[BM][LDP];  // sAp[m][p] = pk(A[m][2p], A[m][2p+1])
  __shared__ unsigned int sBp[DH][LDP];  // sBp[n][p] = pk(W[2p][n], W[2p+1][n])

  const int tid  = threadIdx.x;
  const int lane = tid & 31;
  const int wv   = tid >> 5;
  const int blockRow = blockIdx.x * BM;
  const int mrow = lane & 15;   // M row (A) / N col (B,C) within 16
  const int half = lane >> 4;   // lane group selects K/M half per ISA layout

  v8f acc[8];
#pragma unroll
  for (int t = 0; t < 8; ++t) acc[t] = (v8f){0, 0, 0, 0, 0, 0, 0, 0};

  // ---- register staging buffers (double buffer vs LDS) ----
  float4 aReg[4];
  float2 bReg0[4], bReg1[4];

  auto fetch = [&](int k0) {
#pragma unroll
    for (int i = 0; i < 4; ++i) {
      int idx = tid + i * 256;                    // 0..1023
      // A: 128 rows x 8 float4 per row
      int r = idx >> 3, c = (idx & 7) << 2;
      int grow = blockRow + r;
      aReg[i] = (grow < M) ? *(const float4*)(A + (size_t)grow * K + k0 + c)
                           : make_float4(0.f, 0.f, 0.f, 0.f);
      // B: pair rows (2kp, 2kp+1), two adjacent columns -> coalesced float2
      int npair = idx & 63, kp = idx >> 6;        // n = 2*npair, p = kp
      int n = npair << 1;
      bReg0[i] = *(const float2*)(W + (size_t)(k0 + 2 * kp)     * DH + n);
      bReg1[i] = *(const float2*)(W + (size_t)(k0 + 2 * kp + 1) * DH + n);
    }
  };

  fetch(0);

  for (int k0 = 0; k0 < K; k0 += BK) {
    // ---- store staged chunk to LDS, packed bf16 pairs ----
#pragma unroll
    for (int i = 0; i < 4; ++i) {
      int idx = tid + i * 256;
      int r = idx >> 3, c = (idx & 7) << 2;
      u32x2 ap; ap.x = pk(aReg[i].x, aReg[i].y); ap.y = pk(aReg[i].z, aReg[i].w);
      *(u32x2*)&sAp[r][c >> 1] = ap;
      int npair = idx & 63, kp = idx >> 6;
      int n = npair << 1;
      sBp[n][kp]     = pk(bReg0[i].x, bReg1[i].x);
      sBp[n + 1][kp] = pk(bReg0[i].y, bReg1[i].y);
    }
    __syncthreads();

    // overlap next global fetch with WMMA compute
    if (k0 + BK < K) fetch(k0 + BK);

    // ---- A fragment: pairs {4h..4h+3} and {8+4h..11+4h} -> 2x ds_load_b128 ----
    union Frag { u32x4 q[2]; v16bf v; };
    Frag af;
    const unsigned int* ap = &sAp[wv * 16 + mrow][0];
    af.q[0] = *(const u32x4*)(ap + 4 * half);
    af.q[1] = *(const u32x4*)(ap + 8 + 4 * half);

#pragma unroll
    for (int nt = 0; nt < 8; ++nt) {
      // B fragment: pairs {4h..4h+3} and {8+4h..11+4h} -> 2x ds_load_b128
      Frag bf_;
      const unsigned int* bp = &sBp[nt * 16 + mrow][0];
      bf_.q[0] = *(const u32x4*)(bp + 4 * half);
      bf_.q[1] = *(const u32x4*)(bp + 8 + 4 * half);
      acc[nt] = __builtin_amdgcn_wmma_f32_16x16x32_bf16(
          false, af.v, false, bf_.v, (short)0, acc[nt], false, false);
    }
    __syncthreads();
  }

  // C layout: lanes 0-15 N=lane M=vgpr; lanes 16-31 N=lane-16 M=8+vgpr.
#pragma unroll
  for (int nt = 0; nt < 8; ++nt) {
    int col = nt * 16 + mrow;
    float bv = bias ? bias[col] : 0.0f;
#pragma unroll
    for (int r = 0; r < 8; ++r) {
      int row = blockRow + wv * 16 + r + (half << 3);
      if (row < M) out[(size_t)row * DH + col] = acc[nt][r] + bv;
    }
  }
}

// ---------------- message passing ----------------

__global__ __launch_bounds__(256) void k_zero(float* p, int n_nodes) {
  int idx = blockIdx.x * blockDim.x + threadIdx.x;  // n_nodes*32 float4 slots
  if (idx < n_nodes * 32) ((float4*)p)[idx] = make_float4(0.f, 0.f, 0.f, 0.f);
}

// one wave (32 lanes) per edge; 4 floats per lane
__global__ __launch_bounds__(256) void k_aggregate(const float* __restrict__ hW,
                                                   const int* __restrict__ ei,
                                                   const float* __restrict__ normv,
                                                   float* __restrict__ agg, int E) {
  int idx = blockIdx.x * blockDim.x + threadIdx.x;
  int e = idx >> 5, part = idx & 31;
  if (e >= E) return;
  int s = ei[e], d = ei[E + e];
  float nv = normv[e];
  float4 m = *(const float4*)(hW + (size_t)s * DH + part * 4);
  float* dp = agg + (size_t)d * DH + part * 4;
  atomicAdd(dp + 0, m.x * nv);
  atomicAdd(dp + 1, m.y * nv);
  atomicAdd(dp + 2, m.z * nv);
  atomicAdd(dp + 3, m.w * nv);
}

// h = relu(agg + hW*selfnorm + b) + h   (self-loop folded in, no atomics)
__global__ __launch_bounds__(256) void k_post(float* __restrict__ h,
                                              const float* __restrict__ agg,
                                              const float* __restrict__ hW,
                                              const float* __restrict__ selfn,
                                              const float* __restrict__ bias, int n) {
  int idx = blockIdx.x * blockDim.x + threadIdx.x;
  int i = idx >> 5, part = idx & 31;
  if (i >= n) return;
  float sn = selfn[i];
  size_t off = (size_t)i * DH + part * 4;
  float4 a = *(const float4*)(agg + off);
  float4 w = *(const float4*)(hW + off);
  float4 r = *(const float4*)(h + off);
  int c = part * 4;
  float4 o;
  o.x = fmaxf(a.x + w.x * sn + bias[c + 0], 0.f) + r.x;
  o.y = fmaxf(a.y + w.y * sn + bias[c + 1], 0.f) + r.y;
  o.z = fmaxf(a.z + w.z * sn + bias[c + 2], 0.f) + r.z;
  o.w = fmaxf(a.w + w.w * sn + bias[c + 3], 0.f) + r.w;
  *(float4*)(h + off) = o;
}

// logits[i] = h[i,:] . w_out + b_out : one wave per node, shuffle reduce
__global__ __launch_bounds__(256) void k_output(const float* __restrict__ h,
                                                const float* __restrict__ w_out,
                                                const float* __restrict__ b_out,
                                                float* __restrict__ out, int n) {
  int idx = blockIdx.x * blockDim.x + threadIdx.x;
  int i = idx >> 5, lane = idx & 31;
  if (i >= n) return;
  float4 hv = *(const float4*)(h + (size_t)i * DH + lane * 4);
  float4 wv = *(const float4*)(w_out + lane * 4);
  float s = hv.x * wv.x + hv.y * wv.y + hv.z * wv.z + hv.w * wv.w;
#pragma unroll
  for (int off = 16; off > 0; off >>= 1) s += __shfl_down(s, off, 32);
  if (lane == 0) out[i] = s + b_out[0];
}

// ---------------- launch ----------------

extern "C" void kernel_launch(void* const* d_in, const int* in_sizes, int n_in,
                              void* d_out, int out_size, void* d_ws, size_t ws_size,
                              hipStream_t stream) {
  (void)n_in; (void)out_size; (void)ws_size;
  const float* x       = (const float*)d_in[0];
  const int*   ei      = (const int*)d_in[1];
  const float* ew      = (const float*)d_in[2];
  const float* W_align = (const float*)d_in[4];
  const float* b_align = (const float*)d_in[5];
  const float* W1      = (const float*)d_in[6];
  const float* b1      = (const float*)d_in[7];
  const float* W2      = (const float*)d_in[8];
  const float* b2      = (const float*)d_in[9];
  const float* w_out   = (const float*)d_in[10];
  const float* b_out   = (const float*)d_in[11];
  float* out = (float*)d_out;

  const int N   = in_sizes[3];          // node_type has N elements
  const int E   = in_sizes[2];          // edge_weight has E elements
  const int DIN = in_sizes[0] / N;      // 768

  // workspace carve-up (fp32)
  float* deg   = (float*)d_ws;                                  // N (becomes dinv)
  float* selfn = deg   + (((size_t)N + 63) & ~(size_t)63);      // N
  float* normv = selfn + (((size_t)N + 63) & ~(size_t)63);      // E
  float* h     = normv + (((size_t)E + 63) & ~(size_t)63);      // N*128
  float* hW    = h     + (size_t)N * DH;                        // N*128
  float* agg   = hW    + (size_t)N * DH;                        // N*128

  const int T = 256;
  k_deg_init <<<(N + T - 1) / T, T, 0, stream>>>(deg, N);
  k_deg_accum<<<(E + T - 1) / T, T, 0, stream>>>(ei, ew, deg, E);
  k_dinv     <<<(N + T - 1) / T, T, 0, stream>>>(deg, selfn, N);
  k_norm     <<<(E + T - 1) / T, T, 0, stream>>>(ei, ew, deg, normv, E);

  const int gemmBlocks = (N + BM - 1) / BM;
  k_gemm_wmma_bf16<<<gemmBlocks, 256, 0, stream>>>(x, W_align, b_align, h, N, DIN);

  const int vec = N * 32;   // float4 slots / wave-lane slots over [N,128]
  for (int layer = 0; layer < 2; ++layer) {
    const float* W = layer ? W2 : W1;
    const float* b = layer ? b2 : b1;
    k_gemm_wmma_bf16<<<gemmBlocks, 256, 0, stream>>>(h, W, nullptr, hW, N, DH);
    k_zero     <<<(vec + T - 1) / T, T, 0, stream>>>(agg, N);
    k_aggregate<<<((E * 32) + T - 1) / T, T, 0, stream>>>(hW, ei, normv, agg, E);
    k_post     <<<(vec + T - 1) / T, T, 0, stream>>>(h, agg, hW, selfn, b, N);
  }

  k_output<<<(vec + T - 1) / T, T, 0, stream>>>(h, w_out, b_out, out, N);
}
